// MDTA_57904749085384
// MI455X (gfx1250) — matrix operand order repaired and checked by reading, not compiled, not run on hardware
//
#include <hip/hip_runtime.h>
#include <hip/hip_bf16.h>

#define BATCH 4
#define CH    256
#define HH    192
#define WWD   192
#define NPIX  (HH*WWD)      // 36864
#define HEADS 8
#define HA    98            // db3 analysis output length: floor((192+8-6)/2)+1

typedef __attribute__((ext_vector_type(16))) __bf16 v16bf;
typedef __attribute__((ext_vector_type(8)))  float  v8f;

// analysis correlation kernels (pre-flipped dec filters) and synthesis kernels
__constant__ float cH0A[6] = { 0.33267055295095688f,  0.80689150931333875f,  0.45987750211933132f,
                              -0.13501102001039084f, -0.085441273882241486f, 0.035226291882100656f};
__constant__ float cH1A[6] = { 0.035226291882100656f, 0.085441273882241486f, -0.13501102001039084f,
                              -0.45987750211933132f,  0.80689150931333875f, -0.33267055295095688f};
__constant__ float cG0S[6] = { 0.035226291882100656f, -0.085441273882241486f, -0.13501102001039084f,
                               0.45987750211933132f,   0.80689150931333875f,  0.33267055295095688f};
__constant__ float cG1S[6] = {-0.33267055295095688f,  0.80689150931333875f, -0.45987750211933132f,
                              -0.13501102001039084f,  0.085441273882241486f, 0.035226291882100656f};

// ---------------------------------------------------------------------------
// 1x1 conv as GEMM:  Y[b,o,n] = sum_c W[o,c] * X[b,c,n]   (CI == 256)
// wave tile: M=16 x N=64 (4 accum tiles), block = 8 waves -> M=32 x N=256
// X slabs (K=32 x N=256 fp32) staged via global_load_async_to_lds_b128,
// double-buffered in LDS; ASYNCcnt-pipelined against WMMA.
// ---------------------------------------------------------------------------
__global__ __launch_bounds__(256) void k_gemm1x1(
    const float* __restrict__ Wt,   // (CO, 256)
    const float* __restrict__ X,    // (B, 256, NPIX)
    float* __restrict__ Y,          // (B, CO, NPIX)
    int CO)
{
  __shared__ float tile[2][32 * 256];          // 2 x 32KB double buffer
  const int tid    = threadIdx.x;
  const int lane   = tid & 31;
  const int wv     = tid >> 5;
  const int b      = blockIdx.z;
  const int nBlock = blockIdx.x * 256;
  const int mBase  = blockIdx.y * 32 + (wv & 1) * 16;
  const int nWave  = (wv >> 1) * 64;           // wave's n offset inside block tile
  const float* Xb  = X + (size_t)b * 256 * NPIX;
  float* Yb        = Y + (size_t)b * (size_t)CO * NPIX;

  const int m    = lane & 15;
  const int kh   = (lane >> 4) * 8;            // unused; A and B both offset k by kb16
  const int kb16 = (lane >> 4) * 16;           // B: K offset per lane half
  const int nOff = lane & 15;

  // low 32 bits of a generic __shared__ pointer == wave-relative LDS byte offset
  const unsigned ldsBase = (unsigned)(uintptr_t)&tile[0][0];

  // stage K-slab [kb, kb+32) x [nBlock, nBlock+256) into buffer `buf`
  auto issue_slab = [&](int buf, int kb) {
#pragma unroll
    for (int sub = 0; sub < 8; ++sub) {
      int chunk = sub * 256 + tid;             // 2048 x 16B chunks = 32KB
      int row   = chunk >> 6;                  // 0..31  (K within slab)
      int col   = (chunk & 63) * 4;            // 0..252 (n within block tile)
      unsigned long long gaddr =
          (unsigned long long)(const void*)(Xb + (size_t)(kb + row) * NPIX + (nBlock + col));
      unsigned ldsoff = ldsBase + (unsigned)(buf * (32 * 256 * 4) + chunk * 16);
      asm volatile("global_load_async_to_lds_b128 %0, %1, off"
                   :: "v"(ldsoff), "v"(gaddr) : "memory");
    }
  };

  v8f acc[4] = {v8f{}, v8f{}, v8f{}, v8f{}};

  issue_slab(0, 0);
  for (int kb = 0; kb < 256; kb += 32) {
    const int cur = (kb >> 5) & 1;
    if (kb + 32 < 256) {
      issue_slab(cur ^ 1, kb + 32);
      asm volatile("s_wait_asynccnt 0x8" ::: "memory");  // older 8 (cur slab) done
    } else {
      asm volatile("s_wait_asynccnt 0x0" ::: "memory");
    }
    __syncthreads();                            // copies from all waves visible

    // A fragment (weights, 16x32 bf16) from global/L2
    v16bf a;
    const float* wrow = Wt + (size_t)(mBase + m) * 256;
#pragma unroll
    for (int v = 0; v < 8; ++v) {
      int k0 = kb + kb16 + 2 * v;
      float2 p = *(const float2*)(wrow + k0);
      a[2 * v]     = (__bf16)p.x;
      a[2 * v + 1] = (__bf16)p.y;
    }
    const float* tb = &tile[cur][0];
#pragma unroll
    for (int t = 0; t < 4; ++t) {
      const int col = nWave + t * 16 + nOff;
      v16bf bb;
#pragma unroll
      for (int e = 0; e < 16; ++e)
        bb[e] = (__bf16)tb[(kb16 + e) * 256 + col];
      acc[t] = __builtin_amdgcn_wmma_f32_16x16x32_bf16(false, a, false, bb,
                                                       (short)0, acc[t], false, false);
    }
    __syncthreads();                            // readers done before slab reuse
  }

  const int mo = (lane >> 4) * 8;
#pragma unroll
  for (int t = 0; t < 4; ++t) {
    const int n = nBlock + nWave + t * 16 + nOff;
#pragma unroll
    for (int r = 0; r < 8; ++r)
      Yb[(size_t)(mBase + mo + r) * NPIX + n] = acc[t][r];
  }
}

// ---------------------------------------------------------------------------
// depthwise 3x3, zero pad 1
// ---------------------------------------------------------------------------
__global__ __launch_bounds__(256) void k_dw3x3(
    const float* __restrict__ In, const float* __restrict__ Wd,
    float* __restrict__ Out, int Cc, int Hh, int Ww)
{
  size_t idx = (size_t)blockIdx.x * 256 + threadIdx.x;
  size_t total = (size_t)BATCH * Cc * Hh * Ww;
  if (idx >= total) return;
  int xx = (int)(idx % Ww);
  size_t t1 = idx / Ww;
  int yy = (int)(t1 % Hh);
  size_t bc = t1 / Hh;                 // b*Cc + c
  int c = (int)(bc % Cc);
  const float* w   = Wd + (size_t)c * 9;
  const float* img = In + bc * (size_t)Hh * Ww;
  float s = 0.f;
#pragma unroll
  for (int dy = -1; dy <= 1; ++dy) {
    int y2 = yy + dy;
    if (y2 < 0 || y2 >= Hh) continue;
#pragma unroll
    for (int dx = -1; dx <= 1; ++dx) {
      int x2 = xx + dx;
      if (x2 < 0 || x2 >= Ww) continue;
      s += img[(size_t)y2 * Ww + x2] * w[(dy + 1) * 3 + (dx + 1)];
    }
  }
  Out[idx] = s;
}

// ---------------------------------------------------------------------------
// DWT analysis along W: (B*C,192,192) -> lo,hi (B*C,192,98); stride 2, pad 4
// ---------------------------------------------------------------------------
__global__ __launch_bounds__(256) void k_dwt_rows(
    const float* __restrict__ X, float* __restrict__ Lo, float* __restrict__ Hi)
{
  size_t idx = (size_t)blockIdx.x * 256 + threadIdx.x;   // B*C*192*98 elems
  int ox = (int)(idx % HA);
  size_t t1 = idx / HA;
  int y = (int)(t1 % HH);
  size_t bc = t1 / HH;
  const float* row = X + bc * (size_t)(HH * WWD) + (size_t)y * WWD;
  float s0 = 0.f, s1 = 0.f;
  int xs = 2 * ox - 4;
#pragma unroll
  for (int t = 0; t < 6; ++t) {
    int xi = xs + t;
    if (xi >= 0 && xi < WWD) {
      float v = row[xi];
      s0 += v * cH0A[t];
      s1 += v * cH1A[t];
    }
  }
  Lo[idx] = s0; Hi[idx] = s1;
}

// DWT analysis along H: (B*C,192,98) -> a,b (B*C,98,98)
__global__ __launch_bounds__(256) void k_dwt_cols(
    const float* __restrict__ X, float* __restrict__ Oa, float* __restrict__ Ob)
{
  size_t idx = (size_t)blockIdx.x * 256 + threadIdx.x;   // B*C*98*98 elems
  int xx = (int)(idx % HA);
  size_t t1 = idx / HA;
  int oy = (int)(t1 % HA);
  size_t bc = t1 / HA;
  const float* col = X + bc * (size_t)(HH * HA) + xx;
  float s0 = 0.f, s1 = 0.f;
  int ys = 2 * oy - 4;
#pragma unroll
  for (int t = 0; t < 6; ++t) {
    int yi = ys + t;
    if (yi >= 0 && yi < HH) {
      float v = col[(size_t)yi * HA];
      s0 += v * cH0A[t];
      s1 += v * cH1A[t];
    }
  }
  Oa[idx] = s0; Ob[idx] = s1;
}

// IDWT synthesis along H: a,b (B*C,98,98) -> out (B*C,192,98); lhs_dil 2, pad 1
__global__ __launch_bounds__(256) void k_idwt_cols(
    const float* __restrict__ A, const float* __restrict__ Bm, float* __restrict__ Out)
{
  size_t idx = (size_t)blockIdx.x * 256 + threadIdx.x;   // B*C*192*98 elems
  int xx = (int)(idx % HA);
  size_t t1 = idx / HA;
  int oy = (int)(t1 % HH);
  size_t bc = t1 / HH;
  const float* a = A  + bc * (size_t)(HA * HA);
  const float* b = Bm + bc * (size_t)(HA * HA);
  float s = 0.f;
#pragma unroll
  for (int t = 0; t < 6; ++t) {
    int j = oy + t - 1;                       // index into dilated+padded axis
    if (j >= 0 && j <= 2 * HA - 2 && !(j & 1)) {
      int yi = j >> 1;
      s += a[(size_t)yi * HA + xx] * cG0S[t] + b[(size_t)yi * HA + xx] * cG1S[t];
    }
  }
  Out[idx] = s;
}

// IDWT synthesis along W: a,b (B*C,192,98) -> q (B*C,192,192)
__global__ __launch_bounds__(256) void k_idwt_rows(
    const float* __restrict__ A, const float* __restrict__ Bm, float* __restrict__ Out)
{
  size_t idx = (size_t)blockIdx.x * 256 + threadIdx.x;   // B*C*192*192 elems
  int ox = (int)(idx % WWD);
  size_t t1 = idx / WWD;
  int y = (int)(t1 % HH);
  size_t bc = t1 / HH;
  const float* a = A  + bc * (size_t)(HH * HA) + (size_t)y * HA;
  const float* b = Bm + bc * (size_t)(HH * HA) + (size_t)y * HA;
  float s = 0.f;
#pragma unroll
  for (int t = 0; t < 6; ++t) {
    int j = ox + t - 1;
    if (j >= 0 && j <= 2 * HA - 2 && !(j & 1)) {
      int xi = j >> 1;
      s += a[xi] * cG0S[t] + b[xi] * cG1S[t];
    }
  }
  Out[idx] = s;
}

// ---------------------------------------------------------------------------
// per-row L2 norms of q and k (rows of length NPIX); 2048 rows total
// ---------------------------------------------------------------------------
__global__ __launch_bounds__(256) void k_rownorm(
    const float* __restrict__ Q, const float* __restrict__ KV,
    float* __restrict__ qn, float* __restrict__ kn)
{
  __shared__ float red[256];
  int r = blockIdx.x;
  const float* base;
  float* outp;
  if (r < BATCH * CH) {
    int b = r >> 8, c = r & 255;
    base = Q + ((size_t)b * CH + c) * NPIX;
    outp = qn + r;
  } else {
    int rr = r - BATCH * CH;
    int b = rr >> 8, c = rr & 255;
    base = KV + ((size_t)b * 2 * CH + c) * NPIX;   // k = first CH channels of kv
    outp = kn + rr;
  }
  float s = 0.f;
  for (int i = threadIdx.x; i < NPIX; i += 256) {
    float v = base[i];
    s += v * v;
  }
  red[threadIdx.x] = s;
  __syncthreads();
  for (int st = 128; st > 0; st >>= 1) {
    if (threadIdx.x < st) red[threadIdx.x] += red[threadIdx.x + st];
    __syncthreads();
  }
  if (threadIdx.x == 0) *outp = fmaxf(sqrtf(red[0]), 1e-12f);
}

// ---------------------------------------------------------------------------
// attn = softmax( (q . k^T) * temp / (|q||k|) ) per (b,h): 32x32 with K=NPIX
// one block per (b,h); 8 waves split K; LDS reduce; in-kernel softmax
// ---------------------------------------------------------------------------
__global__ __launch_bounds__(256) void k_attn(
    const float* __restrict__ Q, const float* __restrict__ KV,
    const float* __restrict__ qn, const float* __restrict__ kn,
    const float* __restrict__ temp, float* __restrict__ attnOut)
{
  __shared__ float partial[8][32][32];
  __shared__ float attnS[32][32];
  const int lane = threadIdx.x & 31;
  const int wv   = threadIdx.x >> 5;
  const int bh = blockIdx.x, b = bh >> 3, h = bh & 7;
  const float* qB = Q  + (size_t)b * CH * NPIX     + (size_t)h * 32 * NPIX;
  const float* kB = KV + (size_t)b * 2 * CH * NPIX + (size_t)h * 32 * NPIX;

  const int m    = lane & 15;
  const int kh2  = (lane >> 4) * 8;
  const int kb16 = (lane >> 4) * 16;

  v8f acc[2][2] = {{v8f{}, v8f{}}, {v8f{}, v8f{}}};
  const int kChunk = NPIX / 8;          // 4608
  const int k0w = wv * kChunk;

  for (int kk = k0w; kk < k0w + kChunk; kk += 32) {
    v16bf af[2];
#pragma unroll
    for (int ms = 0; ms < 2; ++ms) {
      const float* row = qB + (size_t)(ms * 16 + m) * NPIX;
#pragma unroll
      for (int v = 0; v < 8; ++v) {
        int kq = kk + kb16 + 2 * v;
        float2 p = *(const float2*)(row + kq);
        af[ms][2 * v]     = (__bf16)p.x;
        af[ms][2 * v + 1] = (__bf16)p.y;
      }
    }
    v16bf bf[2];
#pragma unroll
    for (int ns = 0; ns < 2; ++ns) {
      const float* col = kB + (size_t)(ns * 16 + m) * NPIX;   // row e of k == col e of B
#pragma unroll
      for (int e = 0; e < 16; ++e)
        bf[ns][e] = (__bf16)col[kk + kb16 + e];
    }
#pragma unroll
    for (int ms = 0; ms < 2; ++ms)
#pragma unroll
      for (int ns = 0; ns < 2; ++ns)
        acc[ms][ns] = __builtin_amdgcn_wmma_f32_16x16x32_bf16(false, af[ms], false, bf[ns],
                                                              (short)0, acc[ms][ns], false, false);
  }
  const int mo = (lane >> 4) * 8;
#pragma unroll
  for (int ms = 0; ms < 2; ++ms)
#pragma unroll
    for (int ns = 0; ns < 2; ++ns)
#pragma unroll
      for (int r = 0; r < 8; ++r)
        partial[wv][ms * 16 + mo + r][ns * 16 + (lane & 15)] = acc[ms][ns][r];
  __syncthreads();

  for (int j = threadIdx.x; j < 1024; j += 256) {
    int d = j >> 5, e = j & 31;
    float s = 0.f;
#pragma unroll
    for (int w = 0; w < 8; ++w) s += partial[w][d][e];
    attnS[d][e] = s;
  }
  __syncthreads();

  if (threadIdx.x < 32) {
    int d = threadIdx.x;
    float tq = temp[h] / qn[b * CH + h * 32 + d];
    float row[32];
    float mx = -3.4e38f;
#pragma unroll
    for (int e = 0; e < 32; ++e) {
      float v = attnS[d][e] * tq / kn[b * CH + h * 32 + e];
      row[e] = v;
      mx = fmaxf(mx, v);
    }
    float sum = 0.f;
#pragma unroll
    for (int e = 0; e < 32; ++e) { row[e] = __expf(row[e] - mx); sum += row[e]; }
    float inv = 1.0f / sum;
#pragma unroll
    for (int e = 0; e < 32; ++e)
      attnOut[(size_t)bh * 1024 + d * 32 + e] = row[e] * inv;
  }
}

// ---------------------------------------------------------------------------
// out[b, h*32+d, n] = sum_e attn[b,h,d,e] * v[b,h,e,n]; wave tile 32x64, K=32
// ---------------------------------------------------------------------------
__global__ __launch_bounds__(256) void k_attnv(
    const float* __restrict__ attn, const float* __restrict__ KV,
    float* __restrict__ OutB)
{
  const int lane = threadIdx.x & 31;
  const int wv   = threadIdx.x >> 5;
  const int bh = blockIdx.y, b = bh >> 3, h = bh & 7;
  const int nBase = blockIdx.x * 512 + wv * 64;
  const float* aB = attn + (size_t)bh * 1024;
  const float* vB = KV + (size_t)b * 2 * CH * NPIX + (size_t)(CH + h * 32) * NPIX;
  float* oB       = OutB + (size_t)b * CH * NPIX + (size_t)h * 32 * NPIX;

  const int m    = lane & 15;
  const int kh2  = (lane >> 4) * 8;
  const int kb16 = (lane >> 4) * 16;
  const int mo   = (lane >> 4) * 8;

  v16bf af[2];
#pragma unroll
  for (int ms = 0; ms < 2; ++ms) {
    const float* row = aB + (size_t)(ms * 16 + m) * 32;
#pragma unroll
    for (int v = 0; v < 8; ++v) {
      int e0 = kb16 + 2 * v;
      af[ms][2 * v]     = (__bf16)row[e0];
      af[ms][2 * v + 1] = (__bf16)row[e0 + 1];
    }
  }
#pragma unroll
  for (int t = 0; t < 4; ++t) {
    const int n = nBase + t * 16 + (lane & 15);
    v16bf bf;
#pragma unroll
    for (int e = 0; e < 16; ++e)
      bf[e] = (__bf16)vB[(size_t)(kb16 + e) * NPIX + n];
#pragma unroll
    for (int ms = 0; ms < 2; ++ms) {
      v8f acc = {};
      acc = __builtin_amdgcn_wmma_f32_16x16x32_bf16(false, af[ms], false, bf,
                                                    (short)0, acc, false, false);
#pragma unroll
      for (int r = 0; r < 8; ++r)
        oB[(size_t)(ms * 16 + mo + r) * NPIX + n] = acc[r];
    }
  }
}

// ---------------------------------------------------------------------------
extern "C" void kernel_launch(void* const* d_in, const int* in_sizes, int n_in,
                              void* d_out, int out_size, void* d_ws, size_t ws_size,
                              hipStream_t stream) {
  const float* x       = (const float*)d_in[0];
  const float* qkv_w   = (const float*)d_in[1];   // (512,256)
  const float* qkv_dw  = (const float*)d_in[2];   // (512,1,3,3)
  const float* conv5_w = (const float*)d_in[3];   // (256,1,3,3)
  const float* conv7_w = (const float*)d_in[4];
  const float* conv9_w = (const float*)d_in[5];
  const float* proj_w  = (const float*)d_in[6];   // (256,256)
  const float* temp    = (const float*)d_in[7];   // (8)
  float* out = (float*)d_out;
  float* ws  = (float*)d_ws;

  const size_t NN = (size_t)BATCH * CH * NPIX;     // 37,748,736
  const size_t RN = (size_t)BATCH * CH * HH * HA;  // 19,267,584
  const size_t SN = (size_t)BATCH * CH * HA * HA;  // 9,834,496

  float* kvA   = ws;              // (B,512,N) raw qkv
  float* kvB   = kvA + 2 * NN;    // (B,512,N) after depthwise: [k | v]
  float* rowsL = kvB + 2 * NN;    // (B,C,192,98)  analysis lo rows / reused idwt lo
  float* rowsH = rowsL + RN;      // (B,C,192,98)  analysis hi rows / reused idwt hi
  float* sub   = rowsH + RN;      // 4 x (B,C,98,98): ll,lh,hl,hh
  float* subD  = sub + 4 * SN;    // 4 x (B,C,98,98) after depthwise
  float* qbuf  = subD + 4 * SN;   // (B,C,N)
  float* outB  = qbuf + NN;       // (B,C,N) attn*v
  float* qn    = outB + NN;       // 1024
  float* kn    = qn + BATCH * CH; // 1024
  float* attnW = kn + BATCH * CH; // 32*32*32

  // kv = qkv_w @ x, then depthwise 3x3
  k_gemm1x1<<<dim3(NPIX / 256, 512 / 32, BATCH), 256, 0, stream>>>(qkv_w, x, kvA, 512);
  k_dw3x3<<<(unsigned)((2 * NN) / 256), 256, 0, stream>>>(kvA, qkv_dw, kvB, 512, HH, WWD);

  // wavelet query path
  k_dwt_rows<<<(unsigned)(RN / 256), 256, 0, stream>>>(x, rowsL, rowsH);
  k_dwt_cols<<<(unsigned)(SN / 256), 256, 0, stream>>>(rowsL, sub + 0 * SN, sub + 1 * SN); // ll,lh
  k_dwt_cols<<<(unsigned)(SN / 256), 256, 0, stream>>>(rowsH, sub + 2 * SN, sub + 3 * SN); // hl,hh
  k_dw3x3<<<(unsigned)(SN / 256), 256, 0, stream>>>(sub + 0 * SN, conv5_w, subD + 0 * SN, CH, HA, HA);
  k_dw3x3<<<(unsigned)(SN / 256), 256, 0, stream>>>(sub + 1 * SN, conv5_w, subD + 1 * SN, CH, HA, HA);
  k_dw3x3<<<(unsigned)(SN / 256), 256, 0, stream>>>(sub + 2 * SN, conv7_w, subD + 2 * SN, CH, HA, HA);
  k_dw3x3<<<(unsigned)(SN / 256), 256, 0, stream>>>(sub + 3 * SN, conv9_w, subD + 3 * SN, CH, HA, HA);
  k_idwt_cols<<<(unsigned)(RN / 256), 256, 0, stream>>>(subD + 0 * SN, subD + 1 * SN, rowsL);
  k_idwt_cols<<<(unsigned)(RN / 256), 256, 0, stream>>>(subD + 2 * SN, subD + 3 * SN, rowsH);
  k_idwt_rows<<<(unsigned)(NN / 256), 256, 0, stream>>>(rowsL, rowsH, qbuf);

  // attention
  k_rownorm<<<2 * BATCH * CH, 256, 0, stream>>>(qbuf, kvB, qn, kn);
  k_attn<<<BATCH * HEADS, 256, 0, stream>>>(qbuf, kvB, qn, kn, temp, attnW);
  k_attnv<<<dim3(NPIX / 512, BATCH * HEADS), 256, 0, stream>>>(attnW, kvB, outB);

  // projection
  k_gemm1x1<<<dim3(NPIX / 256, 256 / 32, BATCH), 256, 0, stream>>>(proj_w, outB, out, 256);
}
